// GeMAttention_4475355922480
// MI455X (gfx1250) — compile-verified, matrix operand order checked by reading, not run on hardware
//
#include <hip/hip_runtime.h>
#include <hip/hip_bf16.h>

// ---------------------------------------------------------------------------
// Types
// ---------------------------------------------------------------------------
typedef __attribute__((ext_vector_type(16))) __bf16 v16bf;
typedef __attribute__((ext_vector_type(8)))  float  v8f;
typedef unsigned int v4u __attribute__((ext_vector_type(4)));
typedef int          v4i __attribute__((ext_vector_type(4)));
typedef int          v8i __attribute__((ext_vector_type(8)));

struct U128 { unsigned int x, y, z, w; };          // POD 16-byte vector
union  F16  { U128 u[2]; v16bf v; };               // one bf16 WMMA A/B fragment

#define EPSV      1e-10f
#define DONUT_MIN 0.001f
#define DONUT_MAX 100.0f

#ifdef __gfx1250__
#define HAVE_TDM 1
#else
#define HAVE_TDM 0
#endif

__device__ inline unsigned short f2bf(float f) {
    unsigned u = __float_as_uint(f);
    unsigned r = u + 0x7FFFu + ((u >> 16) & 1u);   // round-to-nearest-even
    return (unsigned short)(r >> 16);
}

__device__ inline float donutf(float x) {
    float s = (x + DONUT_MIN * 0.5f) >= 0.0f ? 1.0f : -1.0f;
    float a = fminf(fmaxf(fabsf(x), DONUT_MIN), DONUT_MAX);
    return s * a;
}

// ---------------------------------------------------------------------------
// Tensor Data Mover: 2-D bf16 tile (tile_d0 x tile_d1 elems, row stride
// stride0 elems in global) -> LDS at lds_addr, optional LDS row padding.
// D# packing per CDNA5 ISA ch.8.3/8.4. One wave issues; EXEC ignored.
// ---------------------------------------------------------------------------
__device__ inline void tdm_load_2d_bf16(unsigned lds_addr, const void* gptr,
                                        unsigned tile_d0, unsigned tile_d1,
                                        unsigned tensor_d0, unsigned tensor_d1,
                                        unsigned long long stride0,
                                        int pad_en, unsigned pad_interval,
                                        unsigned pad_amount)
{
#if HAVE_TDM
    unsigned long long ga = (unsigned long long)gptr;
    v4u g0;
    g0.x = 1u;                                        // count=1, flags=0
    g0.y = lds_addr;                                  // LDS byte address
    g0.z = (unsigned)(ga & 0xFFFFFFFFu);              // global_addr[31:0]
    g0.w = (unsigned)((ga >> 32) & 0x01FFFFFFu)       // global_addr[56:32]
         | (2u << 30);                                // type=2 ("image")
    v8i g1;
    unsigned w0 = (1u << 16)                          // data_size=1 (2 bytes)
                | (pad_en ? (1u << 20) : 0u)
                | ((pad_interval & 7u) << 22)
                | ((pad_amount & 127u) << 25);
    g1[0] = (int)w0;
    g1[1] = (int)((tensor_d0 & 0xFFFFu) << 16);       // abar=0 | t_dim0[15:0]
    g1[2] = (int)(((tensor_d0 >> 16) & 0xFFFFu) | ((tensor_d1 & 0xFFFFu) << 16));
    g1[3] = (int)(((tensor_d1 >> 16) & 0xFFFFu) | ((tile_d0 & 0xFFFFu) << 16));
    g1[4] = (int)(tile_d1 & 0xFFFFu);                 // tile_dim2 = 0
    g1[5] = (int)(unsigned)(stride0 & 0xFFFFFFFFull); // dim0 stride lo
    g1[6] = (int)(unsigned)((stride0 >> 32) & 0xFFFFull); // stride hi | s1=0
    g1[7] = 0;
    v4i z4 = {0, 0, 0, 0};
#if __clang_major__ >= 23
    v8i z8 = {0, 0, 0, 0, 0, 0, 0, 0};
    __builtin_amdgcn_tensor_load_to_lds(g0, g1, z4, z4, z8, 0);
#else
    __builtin_amdgcn_tensor_load_to_lds(g0, g1, z4, z4, 0);
#endif
#else
    (void)lds_addr; (void)gptr; (void)tile_d0; (void)tile_d1;
    (void)tensor_d0; (void)tensor_d1; (void)stride0;
    (void)pad_en; (void)pad_interval; (void)pad_amount;
#endif
}

__device__ inline void wait_tensorcnt0() {
#if HAVE_TDM
    __builtin_amdgcn_s_wait_tensorcnt((short)0);
#endif
}

// ---------------------------------------------------------------------------
// f32 -> bf16 conversion
// ---------------------------------------------------------------------------
__global__ void cvt_f32_bf16(const float* __restrict__ in,
                             unsigned short* __restrict__ out, int n) {
    int i = blockIdx.x * 256 + threadIdx.x;
    if (i < n) out[i] = f2bf(in[i]);
}

// ---------------------------------------------------------------------------
// C[M,N] = A[M,K] @ B[N,K]^T   (A,B bf16; C f32 or bf16)
// Block tile 128x64, BK=32, 256 threads = 8 waves, wave tile 32x32 (2x2 WMMA).
// Global->LDS staging via TDM (wave 0 issues; LDS rows padded by descriptor:
// 16 dwords data + 8 dwords pad = 48-halfword rows).
// ---------------------------------------------------------------------------
template <bool OUT_BF16>
__global__ __launch_bounds__(256) void gemm_nt_bf16(
    const unsigned short* __restrict__ A,
    const unsigned short* __restrict__ B,
    void* __restrict__ C, int M, int N, int K)
{
    constexpr int BM = 128, BN = 64, BK = 32, LDT = 48;   // LDT halfwords (96B)
    __shared__ unsigned short As[BM * LDT];
    __shared__ unsigned short Bs[BN * LDT];

    const int tid  = threadIdx.x;
    const int lane = tid & 31;
    const int wave = tid >> 5;
    const int hlf  = lane >> 4;        // lane half (ISA 16-bit layout)
    const int l15  = lane & 15;
    const int tm0  = blockIdx.y * BM;
    const int tn0  = blockIdx.x * BN;
    const int wm   = (wave >> 1) * 32; // 0,32,64,96
    const int wn   = (wave & 1)  * 32; // 0,32

    v8f acc[2][2];
#pragma unroll
    for (int i = 0; i < 2; ++i)
#pragma unroll
        for (int j = 0; j < 2; ++j)
#pragma unroll
            for (int r = 0; r < 8; ++r) acc[i][j][r] = 0.0f;

    for (int kk = 0; kk < K; kk += BK) {
        __syncthreads();                       // previous compute done
#if HAVE_TDM
        if (wave == 0) {
            // pad_interval=3 (16 dwords), pad_amount=7 (8 dwords): 48hw rows
            tdm_load_2d_bf16((unsigned)(size_t)As,
                             A + (size_t)tm0 * K + kk,
                             BK, BM, (unsigned)K, (unsigned)M,
                             (unsigned long long)K, 1, 3u, 7u);
            tdm_load_2d_bf16((unsigned)(size_t)Bs,
                             B + (size_t)tn0 * K + kk,
                             BK, BN, (unsigned)K, (unsigned)N,
                             (unsigned long long)K, 1, 3u, 7u);
            wait_tensorcnt0();
        }
#else
        {
            int idx = tid;
#pragma unroll
            for (int it = 0; it < 2; ++it, idx += 256) {
                int row = idx >> 2, kc = (idx & 3) * 8;
                U128 d = *(const U128*)(A + (size_t)(tm0 + row) * K + kk + kc);
                *(U128*)(As + row * LDT + kc) = d;
            }
            int row = tid >> 2, kc = (tid & 3) * 8;
            U128 d = *(const U128*)(B + (size_t)(tn0 + row) * K + kk + kc);
            *(U128*)(Bs + row * LDT + kc) = d;
        }
#endif
        __syncthreads();                       // tiles visible to all waves

        F16 af[2], bf[2];
#pragma unroll
        for (int i = 0; i < 2; ++i) {
            const unsigned short* pa = As + (wm + i * 16 + l15) * LDT + hlf * 8;
            af[i].u[0] = *(const U128*)(pa);
            af[i].u[1] = *(const U128*)(pa + 16);
            const unsigned short* pb = Bs + (wn + i * 16 + l15) * LDT + hlf * 8;
            bf[i].u[0] = *(const U128*)(pb);
            bf[i].u[1] = *(const U128*)(pb + 16);
        }
#pragma unroll
        for (int i = 0; i < 2; ++i)
#pragma unroll
            for (int j = 0; j < 2; ++j)
                acc[i][j] = __builtin_amdgcn_wmma_f32_16x16x32_bf16(
                    false, af[i].v, false, bf[j].v, (short)0, acc[i][j],
                    false, false);
    }

#pragma unroll
    for (int i = 0; i < 2; ++i)
#pragma unroll
        for (int j = 0; j < 2; ++j) {
            int rbase = tm0 + wm + i * 16 + hlf * 8;
            int cbase = tn0 + wn + j * 16 + l15;
#pragma unroll
            for (int r = 0; r < 8; ++r) {
                size_t off = (size_t)(rbase + r) * N + cbase;
                float v = acc[i][j][r];
                if (OUT_BF16) ((unsigned short*)C)[off] = f2bf(v);
                else          ((float*)C)[off]          = v;
            }
        }
}

// ---------------------------------------------------------------------------
// z = donut(p)*log(clip(|v+shift|,EPS)); column z-max over sequence;
// v' = exp(z - zmax)  (bf16 out).  One thread owns one (b, channel) column.
// ---------------------------------------------------------------------------
__global__ void vz_prep(const float* __restrict__ vproj,
                        const float* __restrict__ p_param,
                        const float* __restrict__ shift,
                        float* __restrict__ zmax_out,
                        unsigned short* __restrict__ vpr,
                        int B, int Cl, int HD)
{
    int g = blockIdx.x * 256 + threadIdx.x;
    if (g >= B * HD) return;
    int b = g / HD, ch = g % HD;
    float p  = donutf(p_param[ch]);
    float sh = shift[ch];
    const float* base = vproj + (size_t)b * Cl * HD + ch;
    float zm = -__builtin_inff();
    for (int c = 0; c < Cl; ++c) {
        float va = fmaxf(fabsf(base[(size_t)c * HD] + sh), EPSV);
        zm = fmaxf(zm, p * __logf(va));
    }
    zmax_out[g] = zm;
    unsigned short* ob = vpr + (size_t)b * Cl * HD + ch;
    for (int c = 0; c < Cl; ++c) {
        float va = fmaxf(fabsf(base[(size_t)c * HD] + sh), EPSV);
        float z  = p * __logf(va);
        ob[(size_t)c * HD] = f2bf(__expf(z - zm));
    }
}

// ---------------------------------------------------------------------------
// Flash attention + gem transform.
// Grid: (B*NH, S/128). 256 threads = 8 waves; wave owns 16 q-rows.
// K/V tiles staged via TDM; V transposed LDS->LDS for B-fragment reads.
// ---------------------------------------------------------------------------
__global__ __launch_bounds__(256) void flash_gem_attn(
    const unsigned short* __restrict__ Qb,   // [B,S,HD] bf16
    const unsigned short* __restrict__ Kb,   // [B,S,HD] bf16
    const unsigned short* __restrict__ Vb,   // [B,S,HD] bf16 (v')
    const float* __restrict__ zmax,          // [B,HD]
    const float* __restrict__ p_param,       // [HD]
    const float* __restrict__ shift,         // [HD]
    unsigned short* __restrict__ Gout,       // [B,S,HD] bf16 gem
    int B, int S, int HD, int NH)
{
    constexpr int AD = 64;                    // head dim
    constexpr int KT = 64;                    // context tile
    __shared__ unsigned short Klds[KT * AD];        // [c][a]
    __shared__ unsigned short Vlds[KT * AD];        // [c][a] (TDM dest)
    __shared__ unsigned short VldsT[AD * KT];       // [a][c]
    __shared__ unsigned short Plds[8][16 * KT];     // per-wave P tile

    const int tid  = threadIdx.x;
    const int lane = tid & 31, wave = tid >> 5;
    const int hlf  = lane >> 4, l15 = lane & 15;
    const int b = blockIdx.x / NH, h = blockIdx.x % NH;
    const int q0 = blockIdx.y * 128 + wave * 16;
    const int chb = h * AD;
    const float scale = rsqrtf((float)HD);    // reference scales by 1/sqrt(hd)

    // Q fragments for this wave's 16 rows (2 chunks over head dim)
    F16 qf[2];
    {
        const unsigned short* qp =
            Qb + (size_t)(b * S + q0 + l15) * HD + chb + hlf * 8;
#pragma unroll
        for (int c2 = 0; c2 < 2; ++c2) {
            qf[c2].u[0] = *(const U128*)(qp + c2 * 32);
            qf[c2].u[1] = *(const U128*)(qp + c2 * 32 + 16);
        }
    }

    float m_i[8], l_i[8];
    v8f o[4];
#pragma unroll
    for (int r = 0; r < 8; ++r) { m_i[r] = -__builtin_inff(); l_i[r] = 0.0f; }
#pragma unroll
    for (int nt = 0; nt < 4; ++nt)
#pragma unroll
        for (int r = 0; r < 8; ++r) o[nt][r] = 0.0f;

    for (int kb = 0; kb < S; kb += KT) {
        __syncthreads();                       // prev compute done
#if HAVE_TDM
        if (wave == 0) {
            tdm_load_2d_bf16((unsigned)(size_t)Klds,
                             Kb + (size_t)(b * S + kb) * HD + chb,
                             AD, KT, (unsigned)HD, (unsigned)(S - kb),
                             (unsigned long long)HD, 0, 0u, 0u);
            tdm_load_2d_bf16((unsigned)(size_t)Vlds,
                             Vb + (size_t)(b * S + kb) * HD + chb,
                             AD, KT, (unsigned)HD, (unsigned)(S - kb),
                             (unsigned long long)HD, 0, 0u, 0u);
            wait_tensorcnt0();
        }
        __syncthreads();                       // Klds/Vlds ready
        for (int idx = tid; idx < KT * AD; idx += 256) {
            int c = idx >> 6, a = idx & 63;
            VldsT[a * KT + c] = Vlds[c * AD + a];
        }
#else
        for (int idx = tid; idx < KT * AD / 8; idx += 256) {
            int c = idx >> 3, a = (idx & 7) * 8;
            U128 d = *(const U128*)(Kb + (size_t)(b * S + kb + c) * HD + chb + a);
            *(U128*)(Klds + c * AD + a) = d;
        }
        for (int idx = tid; idx < KT * AD; idx += 256) {
            int c = idx >> 6, a = idx & 63;
            VldsT[a * KT + c] = Vb[(size_t)(b * S + kb + c) * HD + chb + a];
        }
#endif
        __syncthreads();                       // all tiles ready

        // S = Q @ K^T (4 column tiles of 16)
        v8f s[4];
#pragma unroll
        for (int nt = 0; nt < 4; ++nt) {
            v8f z;
#pragma unroll
            for (int r = 0; r < 8; ++r) z[r] = 0.0f;
#pragma unroll
            for (int c2 = 0; c2 < 2; ++c2) {
                F16 kf;
                const unsigned short* kp =
                    Klds + (nt * 16 + l15) * AD + c2 * 32 + hlf * 8;
                kf.u[0] = *(const U128*)(kp);
                kf.u[1] = *(const U128*)(kp + 16);
                z = __builtin_amdgcn_wmma_f32_16x16x32_bf16(
                        false, qf[c2].v, false, kf.v, (short)0, z, false, false);
            }
            s[nt] = z;
        }

        // online softmax: row max / sum across N (lanes 0..15 within each half)
#pragma unroll
        for (int r = 0; r < 8; ++r) {
            float mx = -__builtin_inff();
#pragma unroll
            for (int nt = 0; nt < 4; ++nt) {
                s[nt][r] *= scale;
                mx = fmaxf(mx, s[nt][r]);
            }
            mx = fmaxf(mx, __shfl_xor(mx, 1, 16));
            mx = fmaxf(mx, __shfl_xor(mx, 2, 16));
            mx = fmaxf(mx, __shfl_xor(mx, 4, 16));
            mx = fmaxf(mx, __shfl_xor(mx, 8, 16));

            float mi    = fmaxf(m_i[r], mx);
            float alpha = __expf(m_i[r] - mi);
            float rs = 0.0f;
#pragma unroll
            for (int nt = 0; nt < 4; ++nt) {
                float pe = __expf(s[nt][r] - mi);
                s[nt][r] = pe;
                rs += pe;
            }
            rs += __shfl_xor(rs, 1, 16);
            rs += __shfl_xor(rs, 2, 16);
            rs += __shfl_xor(rs, 4, 16);
            rs += __shfl_xor(rs, 8, 16);
            l_i[r] = l_i[r] * alpha + rs;
            m_i[r] = mi;
#pragma unroll
            for (int nt = 0; nt < 4; ++nt) o[nt][r] *= alpha;
        }

        // P: C-layout -> A-layout via per-wave LDS bounce (in-order per wave)
#pragma unroll
        for (int nt = 0; nt < 4; ++nt)
#pragma unroll
            for (int r = 0; r < 8; ++r)
                Plds[wave][(r + hlf * 8) * KT + nt * 16 + l15] = f2bf(s[nt][r]);

        // O += P @ V
#pragma unroll
        for (int c2 = 0; c2 < 2; ++c2) {
            F16 pf;
            const unsigned short* pp =
                &Plds[wave][l15 * KT + c2 * 32 + hlf * 8];
            pf.u[0] = *(const U128*)(pp);
            pf.u[1] = *(const U128*)(pp + 16);
#pragma unroll
            for (int nt = 0; nt < 4; ++nt) {
                F16 vf;
                const unsigned short* vp =
                    VldsT + (nt * 16 + l15) * KT + c2 * 32 + hlf * 8;
                vf.u[0] = *(const U128*)(vp);
                vf.u[1] = *(const U128*)(vp + 16);
                o[nt] = __builtin_amdgcn_wmma_f32_16x16x32_bf16(
                            false, pf.v, false, vf.v, (short)0, o[nt],
                            false, false);
            }
        }
    }

    // epilogue: normalize + gem transform, emit bf16
#pragma unroll
    for (int nt = 0; nt < 4; ++nt) {
        int ch = chb + nt * 16 + l15;
        float p  = donutf(p_param[ch]);
        float sh = shift[ch];
        float zm = zmax[b * HD + ch];
#pragma unroll
        for (int r = 0; r < 8; ++r) {
            float attn = o[nt][r] / l_i[r];
            float gem  = __expf((zm + __logf(attn)) / p) - sh;
            int q = q0 + r + hlf * 8;
            Gout[(size_t)(b * S + q) * HD + ch] = f2bf(gem);
        }
    }
}

// ---------------------------------------------------------------------------
// Host orchestration
// ---------------------------------------------------------------------------
extern "C" void kernel_launch(void* const* d_in, const int* in_sizes, int n_in,
                              void* d_out, int out_size, void* d_ws, size_t ws_size,
                              hipStream_t stream) {
    (void)in_sizes; (void)n_in; (void)out_size; (void)ws_size;
    const int B = 2, S = 2048, HD = 1024, NH = 16;
    const int M = B * S;                       // 4096

    const float* query   = (const float*)d_in[0];
    const float* context = (const float*)d_in[1];
    const float* WQ      = (const float*)d_in[2];
    const float* WK      = (const float*)d_in[3];
    const float* WV      = (const float*)d_in[4];
    const float* Wout    = (const float*)d_in[5];
    const float* p_param = (const float*)d_in[6];
    const float* shift   = (const float*)d_in[7];

    unsigned char* ws = (unsigned char*)d_ws;
    size_t off = 0;
    auto carve = [&](size_t bytes) -> void* {
        void* p = ws + off;
        off = (off + bytes + 255) & ~(size_t)255;
        return p;
    };
    unsigned short* query_bf   = (unsigned short*)carve((size_t)M * HD * 2);
    unsigned short* context_bf = (unsigned short*)carve((size_t)M * HD * 2);
    unsigned short* WQb  = (unsigned short*)carve((size_t)HD * HD * 2);
    unsigned short* WKb  = (unsigned short*)carve((size_t)HD * HD * 2);
    unsigned short* WVb  = (unsigned short*)carve((size_t)HD * HD * 2);
    unsigned short* Wob  = (unsigned short*)carve((size_t)HD * HD * 2);
    unsigned short* qbf  = (unsigned short*)carve((size_t)M * HD * 2);
    unsigned short* kbf  = (unsigned short*)carve((size_t)M * HD * 2);
    float*          vprj = (float*)carve((size_t)M * HD * 4);
    unsigned short* vpr  = (unsigned short*)carve((size_t)M * HD * 2);
    float*          zmx  = (float*)carve((size_t)B * HD * 4);
    unsigned short* gemb = (unsigned short*)carve((size_t)M * HD * 2);

    // 1. conversions
    int nact = M * HD, nwt = HD * HD;
    cvt_f32_bf16<<<(nact + 255) / 256, 256, 0, stream>>>(query,   query_bf,   nact);
    cvt_f32_bf16<<<(nact + 255) / 256, 256, 0, stream>>>(context, context_bf, nact);
    cvt_f32_bf16<<<(nwt  + 255) / 256, 256, 0, stream>>>(WQ,   WQb, nwt);
    cvt_f32_bf16<<<(nwt  + 255) / 256, 256, 0, stream>>>(WK,   WKb, nwt);
    cvt_f32_bf16<<<(nwt  + 255) / 256, 256, 0, stream>>>(WV,   WVb, nwt);
    cvt_f32_bf16<<<(nwt  + 255) / 256, 256, 0, stream>>>(Wout, Wob, nwt);

    // 2. projections
    dim3 gb(HD / 64, M / 128);
    gemm_nt_bf16<true ><<<gb, 256, 0, stream>>>(query_bf,   WQb, qbf,  M, HD, HD);
    gemm_nt_bf16<true ><<<gb, 256, 0, stream>>>(context_bf, WKb, kbf,  M, HD, HD);
    gemm_nt_bf16<false><<<gb, 256, 0, stream>>>(context_bf, WVb, vprj, M, HD, HD);

    // 3. v-channel log/exp preprocessing + z-max
    vz_prep<<<(B * HD + 255) / 256, 256, 0, stream>>>(vprj, p_param, shift,
                                                      zmx, vpr, B, S, HD);

    // 4. flash attention + gem transform
    dim3 ga(B * NH, S / 128);
    flash_gem_attn<<<ga, 256, 0, stream>>>(qbf, kbf, vpr, zmx, p_param, shift,
                                           gemb, B, S, HD, NH);

    // 5. output projection (f32 out)
    gemm_nt_bf16<false><<<gb, 256, 0, stream>>>(gemb, Wob, d_out, M, HD, HD);
}
